// LLMModel_48825188221558
// MI455X (gfx1250) — matrix operand order repaired and checked by reading, not compiled, not run on hardware
//
#include <hip/hip_runtime.h>
#include <hip/hip_bf16.h>

// ---------------------------------------------------------------------------
// GPT-2 forward for MI455X (gfx1250, wave32): bf16 WMMA + async global->LDS.
// ---------------------------------------------------------------------------

typedef __bf16 bf16_t;
typedef __attribute__((ext_vector_type(16))) __bf16 bf16x16;
typedef __attribute__((ext_vector_type(8)))  __bf16 bf16x8;
typedef __attribute__((ext_vector_type(8)))  float  f32x8;
typedef int v4i __attribute__((vector_size(4 * sizeof(int))));

union Frag16 { bf16x16 v; bf16x8 h[2]; };

constexpr int cV  = 50257;
constexpr int cH  = 768;
constexpr int cL  = 12;
constexpr int cNH = 12;
constexpr int cI  = 3072;
constexpr int cB  = 2;
constexpr int cS  = 1024;
constexpr int cHD = 64;
constexpr int TOK = cB * cS;   // 2048 rows

static __device__ __forceinline__ f32x8 wmma_bf16(const Frag16& a, const Frag16& b, f32x8 c) {
  return __builtin_amdgcn_wmma_f32_16x16x32_bf16(false, a.v, false, b.v, (short)0, c,
                                                 false, false);
}

// ---- async global -> LDS copy (CDNA5 GLOBAL_LOAD_ASYNC_TO_LDS_B128) --------
// Probe-derived signature: (v4i AS(1)*, v4i AS(3)*, imm offset, imm cpol)
#if defined(__has_builtin)
#if __has_builtin(__builtin_amdgcn_global_load_async_to_lds_b128)
#define HAS_ASYNC_LDS 1
#endif
#endif
#ifndef HAS_ASYNC_LDS
#define HAS_ASYNC_LDS 0
#endif

static __device__ __forceinline__ void cp16(const bf16_t* __restrict__ g,
                                            bf16_t* __restrict__ l) {
#if HAS_ASYNC_LDS
  __builtin_amdgcn_global_load_async_to_lds_b128(
      (__attribute__((address_space(1))) v4i*)g,
      (__attribute__((address_space(3))) v4i*)l, 0, 0);
#else
  *(uint4*)l = *(const uint4*)g;
#endif
}

static __device__ __forceinline__ void async_wait() {
#if HAS_ASYNC_LDS
#if __has_builtin(__builtin_amdgcn_s_wait_asynccnt)
  __builtin_amdgcn_s_wait_asynccnt(0);
#else
  asm volatile("s_wait_asynccnt 0" ::: "memory");
#endif
#endif
}

// ---------------------------------------------------------------------------
// fp32 -> bf16 weight conversion (grid-stride)
// ---------------------------------------------------------------------------
__global__ void cvt_f32_bf16(const float* __restrict__ in, bf16_t* __restrict__ out, size_t n) {
  size_t i = (size_t)blockIdx.x * blockDim.x + threadIdx.x;
  const size_t stride = (size_t)gridDim.x * blockDim.x;
  for (; i < n; i += stride) out[i] = (bf16_t)in[i];
}

// ---------------------------------------------------------------------------
// Embedding: x[t, :] = tok_emb[ids[t], :] + pos_emb[t % S, :]   (fp32)
// ---------------------------------------------------------------------------
__global__ __launch_bounds__(256) void embed_kernel(
    const int* __restrict__ ids, const float* __restrict__ tok,
    const float* __restrict__ pos, float* __restrict__ x)
{
  const int t  = blockIdx.x;
  const int s  = t % cS;
  const int id = ids[t];
#pragma unroll
  for (int j = 0; j < 3; ++j) {
    int i = threadIdx.x + 256 * j;
    x[(size_t)t * cH + i] = tok[(size_t)id * cH + i] + pos[(size_t)s * cH + i];
  }
}

// ---------------------------------------------------------------------------
// LayerNorm (fp32 in) fused with bf16 activation conversion. 1 block / row.
// ---------------------------------------------------------------------------
__global__ __launch_bounds__(256) void ln_to_bf16(
    const float* __restrict__ x, const float* __restrict__ w,
    const float* __restrict__ bb, bf16_t* __restrict__ out)
{
  const int row = blockIdx.x;
  const int tid = threadIdx.x;
  const float* xr = x + (size_t)row * cH;
  float vals[3];
  float s = 0.f, s2 = 0.f;
#pragma unroll
  for (int j = 0; j < 3; ++j) {
    float v = xr[tid + 256 * j];
    vals[j] = v; s += v; s2 += v * v;
  }
#pragma unroll
  for (int o = 16; o > 0; o >>= 1) {
    s  += __shfl_down(s,  o, 32);
    s2 += __shfl_down(s2, o, 32);
  }
  __shared__ float rs[8], rs2[8];
  if ((tid & 31) == 0) { rs[tid >> 5] = s; rs2[tid >> 5] = s2; }
  __syncthreads();
  float tot = 0.f, tot2 = 0.f;
#pragma unroll
  for (int i = 0; i < 8; ++i) { tot += rs[i]; tot2 += rs2[i]; }
  const float mean = tot * (1.0f / cH);
  const float var  = tot2 * (1.0f / cH) - mean * mean;
  const float rstd = rsqrtf(var + 1e-5f);
  bf16_t* orow = out + (size_t)row * cH;
#pragma unroll
  for (int j = 0; j < 3; ++j) {
    int i = tid + 256 * j;
    orow[i] = (bf16_t)((vals[j] - mean) * rstd * w[i] + bb[i]);
  }
}

// ---------------------------------------------------------------------------
// GEMM: D[M,N] = A[M,K] @ W[N,K]^T (+bias)(+GELU)(+resid), bf16 WMMA.
// Block tile 128x64, BK=64, double-buffered async LDS staging.
// 8 waves as 4(M) x 2(N); each wave owns a 32x32 sub-tile (2x2 WMMA accs).
// ---------------------------------------------------------------------------
#define BM 128
#define BN 64
#define BK 64
#define ST 72   // padded LDS row stride (bf16): 144B, 16B-aligned, conflict-free

__global__ __launch_bounds__(256) void gemm_bf16_wmma(
    const bf16_t* __restrict__ A, const bf16_t* __restrict__ W,
    const float* __restrict__ bias, const float* __restrict__ resid,
    bf16_t* __restrict__ out16, float* __restrict__ out32,
    int M, int N, int K, int gelu)
{
  __shared__ bf16_t sA[2][BM * ST];
  __shared__ bf16_t sB[2][BN * ST];
  const int tid   = threadIdx.x;
  const int wave  = tid >> 5;
  const int lane  = tid & 31;
  const int l16   = lane & 15;
  const int half  = lane >> 4;
  const int mwave = wave >> 1;   // 0..3
  const int nwave = wave & 1;    // 0..1
  const int bm = blockIdx.x * BM;
  const int bn = blockIdx.y * BN;

  f32x8 acc[2][2] = {};

  const int nt = K / BK;

  auto stage = [&](int buf, int k0) {
    // A tile 128x64: 1024 x 16B, 4 per thread
#pragma unroll
    for (int i = 0; i < 4; ++i) {
      int idx = tid + i * 256;
      int r = idx >> 3;
      int c = (idx & 7) * 8;
      cp16(&A[(size_t)(bm + r) * K + k0 + c], &sA[buf][r * ST + c]);
    }
    // B tile 64x64: 512 x 16B, 2 per thread (zero-fill rows beyond N)
#pragma unroll
    for (int i = 0; i < 2; ++i) {
      int idx = tid + i * 256;
      int r = idx >> 3;
      int c = (idx & 7) * 8;
      int n = bn + r;
      if (n < N) {
        cp16(&W[(size_t)n * K + k0 + c], &sB[buf][r * ST + c]);
      } else {
        uint4 z = {0, 0, 0, 0};
        *(uint4*)&sB[buf][r * ST + c] = z;
      }
    }
  };

  stage(0, 0);
  async_wait();
  __syncthreads();

  for (int t = 0; t < nt; ++t) {
    if (t + 1 < nt) stage((t + 1) & 1, (t + 1) * BK);   // overlaps WMMAs below

    const bf16_t* a   = sA[t & 1];
    const bf16_t* bsh = sB[t & 1];
#pragma unroll
    for (int ks = 0; ks < BK; ks += 32) {
      Frag16 fa[2], fb[2];
#pragma unroll
      for (int mi = 0; mi < 2; ++mi) {
        int row = 32 * mwave + 16 * mi + l16;
        fa[mi].h[0] = *(const bf16x8*)&a[row * ST + ks + 8 * half];
        fa[mi].h[1] = *(const bf16x8*)&a[row * ST + ks + 8 * half + 16];
      }
#pragma unroll
      for (int ni = 0; ni < 2; ++ni) {
        int row = 32 * nwave + 16 * ni + l16;
        fb[ni].h[0] = *(const bf16x8*)&bsh[row * ST + ks + 8 * half];
        fb[ni].h[1] = *(const bf16x8*)&bsh[row * ST + ks + 8 * half + 16];
      }
#pragma unroll
      for (int mi = 0; mi < 2; ++mi)
#pragma unroll
        for (int ni = 0; ni < 2; ++ni)
          acc[mi][ni] = wmma_bf16(fa[mi], fb[ni], acc[mi][ni]);
    }
    async_wait();      // next tile's async copies done (they ran under the WMMAs)
    __syncthreads();
  }

  // Epilogue. C layout: VGPR r -> (m = r + 8*half, n = l16) within 16x16 tile.
#pragma unroll
  for (int mi = 0; mi < 2; ++mi) {
#pragma unroll
    for (int ni = 0; ni < 2; ++ni) {
#pragma unroll
      for (int r = 0; r < 8; ++r) {
        int m = bm + 32 * mwave + 16 * mi + r + 8 * half;
        int n = bn + 32 * nwave + 16 * ni + l16;
        if (n < N) {
          float v = acc[mi][ni][r];
          if (bias)  v += bias[n];
          if (gelu)  v = 0.5f * v * (1.0f + erff(v * 0.70710678118654752f));
          if (resid) v += resid[(size_t)m * N + n];
          if (out32) out32[(size_t)m * N + n] = v;
          else       out16[(size_t)m * N + n] = (bf16_t)v;
        }
      }
    }
  }
}

// ---------------------------------------------------------------------------
// Flash attention (causal). S^T = K@Q^T keeps the softmax reduction in-register
// and makes the S^T C-fragment layout identical to the P A-fragment layout for
// the P@V WMMA. 8 waves/block (one 16-query tile each) = 128 queries of one
// (b, head). K chunk staged async; V transposed into LDS for the B operand.
// ---------------------------------------------------------------------------
#define SKST 72
#define SVST 40

__global__ __launch_bounds__(256) void attn_flash_wmma(
    const bf16_t* __restrict__ Q, const bf16_t* __restrict__ Kb,
    const bf16_t* __restrict__ Vb, bf16_t* __restrict__ O)
{
  __shared__ bf16_t sK[32 * SKST];   // [key 0..31][hd 0..63]
  __shared__ bf16_t sVT[64 * SVST];  // [hd 0..63][key 0..31]
  const int tid  = threadIdx.x;
  const int wave = tid >> 5;
  const int lane = tid & 31;
  const int l16  = lane & 15;
  const int half = lane >> 4;

  const int nqt = cS / 128;
  const int bh  = blockIdx.x / nqt;
  const int qt  = blockIdx.x % nqt;
  const int b   = bh / cNH;
  const int h   = bh % cNH;
  const int qblk = qt * 128;
  const int qrow = qblk + wave * 16;

  // Q as B operand: lane column = query qrow+l16; frag f covers hd [32f,32f+32)
  Frag16 fq[2];
  {
    const bf16_t* qp = Q + ((size_t)(b * cS) + qrow + l16) * cH + h * cHD;
#pragma unroll
    for (int f = 0; f < 2; ++f) {
      int hd0 = 32 * f + 8 * half;
      fq[f].h[0] = *(const bf16x8*)&qp[hd0];
      fq[f].h[1] = *(const bf16x8*)&qp[hd0 + 16];
    }
  }

  float m_run = -1e30f, l_run = 0.0f;
  f32x8 acc[4] = {};
  const float scale = 0.125f;              // 1/sqrt(64)

  const int kv_end = qblk + 128;
  for (int c = 0; c < kv_end; c += 32) {
    __syncthreads();
    {
      int r  = tid >> 3;
      int cc = (tid & 7) * 8;
      const size_t src = ((size_t)(b * cS) + c + r) * cH + h * cHD + cc;
      cp16(&Kb[src], &sK[r * SKST + cc]);          // async K stage
      uint4 vv = *(const uint4*)&Vb[src];          // V needs transpose: via VGPR
      const bf16_t* pv = (const bf16_t*)&vv;
#pragma unroll
      for (int j = 0; j < 8; ++j) sVT[(cc + j) * SVST + r] = pv[j];
    }
    async_wait();
    __syncthreads();

    // S^T tiles: s0 = keys [c,c+16), s1 = keys [c+16,c+32); A = K rows
    f32x8 s0 = {}, s1 = {};
#pragma unroll
    for (int f = 0; f < 2; ++f) {
      Frag16 a0, a1;
      int ak = 32 * f + 8 * half;
      a0.h[0] = *(const bf16x8*)&sK[l16 * SKST + ak];
      a0.h[1] = *(const bf16x8*)&sK[l16 * SKST + ak + 16];
      a1.h[0] = *(const bf16x8*)&sK[(16 + l16) * SKST + ak];
      a1.h[1] = *(const bf16x8*)&sK[(16 + l16) * SKST + ak + 16];
      s0 = wmma_bf16(a0, fq[f], s0);
      s1 = wmma_bf16(a1, fq[f], s1);
    }

    // scale + causal mask. S^T layout: lane query = l16, key = r + 8*half (+16 for s1)
    const int qg = qrow + l16;
    float cmax = -1e30f;
#pragma unroll
    for (int r = 0; r < 8; ++r) {
      int kg0 = c + r + 8 * half;
      int kg1 = c + 16 + r + 8 * half;
      float v0 = (kg0 <= qg) ? s0[r] * scale : -1e9f;
      float v1 = (kg1 <= qg) ? s1[r] * scale : -1e9f;
      s0[r] = v0; s1[r] = v1;
      cmax = fmaxf(cmax, fmaxf(v0, v1));
    }
    cmax = fmaxf(cmax, __shfl_xor(cmax, 16, 32));
    float m_new = fmaxf(m_run, cmax);
    float alpha = __expf(m_run - m_new);
    m_run = m_new;

    // exponentiate -> P in A-fragment layout directly
    Frag16 pa;
    float psum = 0.0f;
#pragma unroll
    for (int r = 0; r < 8; ++r) {
      float p0 = __expf(s0[r] - m_new);
      float p1 = __expf(s1[r] - m_new);
      psum += p0 + p1;
      pa.v[r]     = (bf16_t)p0;
      pa.v[r + 8] = (bf16_t)p1;
    }
    l_run = l_run * alpha + psum;

    // broadcast per-query alpha into P@V C-layout (queries r + 8*half per lane)
    float ar[8];
#pragma unroll
    for (int r = 0; r < 8; ++r) ar[r] = __shfl(alpha, r + 8 * half, 32);

#pragma unroll
    for (int nf = 0; nf < 4; ++nf) {
#pragma unroll
      for (int r = 0; r < 8; ++r) acc[nf][r] *= ar[r];
      Frag16 fv;                              // B operand = V^T rows (hd), keys contiguous
      int row = nf * 16 + l16;
      int kb  = 8 * half;
      fv.h[0] = *(const bf16x8*)&sVT[row * SVST + kb];
      fv.h[1] = *(const bf16x8*)&sVT[row * SVST + kb + 16];
      acc[nf] = wmma_bf16(pa, fv, acc[nf]);
    }
  }

  float l_tot = l_run + __shfl_xor(l_run, 16, 32);
  float lr[8];
#pragma unroll
  for (int r = 0; r < 8; ++r) lr[r] = 1.0f / __shfl(l_tot, r + 8 * half, 32);

#pragma unroll
  for (int nf = 0; nf < 4; ++nf) {
#pragma unroll
    for (int r = 0; r < 8; ++r) {
      int q  = qrow + r + 8 * half;
      int hd = nf * 16 + l16;
      O[((size_t)(b * cS) + q) * cH + h * cHD + hd] = (bf16_t)(acc[nf][r] * lr[r]);
    }
  }
}

// ---------------------------------------------------------------------------
// Host orchestration
// ---------------------------------------------------------------------------
extern "C" void kernel_launch(void* const* d_in, const int* in_sizes, int n_in,
                              void* d_out, int out_size, void* d_ws, size_t ws_size,
                              hipStream_t stream)
{
  (void)in_sizes; (void)n_in; (void)out_size; (void)ws_size;

  const int*   ids     = (const int*)d_in[0];
  // d_in[1] = attention_mask: all-ones; causal mask applied in-kernel.
  const float* tok_emb = (const float*)d_in[2];
  const float* pos_emb = (const float*)d_in[3];
  const float* ln1_w   = (const float*)d_in[4];
  const float* ln1_b   = (const float*)d_in[5];
  const float* qw      = (const float*)d_in[6];
  const float* kw      = (const float*)d_in[7];
  const float* vw      = (const float*)d_in[8];
  const float* ow      = (const float*)d_in[9];
  const float* ob      = (const float*)d_in[10];
  const float* ln2_w   = (const float*)d_in[11];
  const float* ln2_b   = (const float*)d_in[12];
  const float* w1      = (const float*)d_in[13];
  const float* b1      = (const float*)d_in[14];
  const float* w2      = (const float*)d_in[15];
  const float* b2      = (const float*)d_in[16];
  const float* lnf_w   = (const float*)d_in[17];
  const float* lnf_b   = (const float*)d_in[18];
  const float* lm_w    = (const float*)d_in[19];

  const size_t HH = (size_t)cH * cH;
  const size_t IH = (size_t)cI * cH;

  char* wsp = (char*)d_ws;
  auto alloc = [&](size_t bytes) -> void* {
    void* p = (void*)wsp;
    wsp += (bytes + 255) & ~(size_t)255;
    return p;
  };
  bf16_t* qw16 = (bf16_t*)alloc(cL * HH * sizeof(bf16_t));
  bf16_t* kw16 = (bf16_t*)alloc(cL * HH * sizeof(bf16_t));
  bf16_t* vw16 = (bf16_t*)alloc(cL * HH * sizeof(bf16_t));
  bf16_t* ow16 = (bf16_t*)alloc(cL * HH * sizeof(bf16_t));
  bf16_t* w116 = (bf16_t*)alloc(cL * IH * sizeof(bf16_t));
  bf16_t* w216 = (bf16_t*)alloc(cL * IH * sizeof(bf16_t));
  bf16_t* lm16 = (bf16_t*)alloc((size_t)cV * cH * sizeof(bf16_t));
  float*  x    = (float*) alloc((size_t)TOK * cH * sizeof(float));
  float*  x2   = (float*) alloc((size_t)TOK * cH * sizeof(float));
  bf16_t* h16  = (bf16_t*)alloc((size_t)TOK * cH * sizeof(bf16_t));
  bf16_t* q16  = (bf16_t*)alloc((size_t)TOK * cH * sizeof(bf16_t));
  bf16_t* k16  = (bf16_t*)alloc((size_t)TOK * cH * sizeof(bf16_t));
  bf16_t* v16b = (bf16_t*)alloc((size_t)TOK * cH * sizeof(bf16_t));
  bf16_t* a16  = (bf16_t*)alloc((size_t)TOK * cH * sizeof(bf16_t));
  bf16_t* ff16 = (bf16_t*)alloc((size_t)TOK * cI * sizeof(bf16_t));

  auto cvt = [&](const float* src, bf16_t* dst, size_t n) {
    cvt_f32_bf16<<<2048, 256, 0, stream>>>(src, dst, n);
  };
  auto gemm = [&](const bf16_t* Am, const bf16_t* Wm, const float* bias,
                  const float* resid, bf16_t* o16, float* o32,
                  int M, int N, int K, int gelu) {
    dim3 grid(M / BM, (N + BN - 1) / BN);
    gemm_bf16_wmma<<<grid, dim3(256), 0, stream>>>(Am, Wm, bias, resid, o16, o32,
                                                   M, N, K, gelu);
  };

  cvt(qw, qw16, cL * HH);
  cvt(kw, kw16, cL * HH);
  cvt(vw, vw16, cL * HH);
  cvt(ow, ow16, cL * HH);
  cvt(w1, w116, cL * IH);
  cvt(w2, w216, cL * IH);
  cvt(lm_w, lm16, (size_t)cV * cH);

  embed_kernel<<<TOK, 256, 0, stream>>>(ids, tok_emb, pos_emb, x);

  float* xa = x;
  float* xb = x2;
  for (int l = 0; l < cL; ++l) {
    ln_to_bf16<<<TOK, 256, 0, stream>>>(xa, ln1_w + (size_t)l * cH, ln1_b + (size_t)l * cH, h16);
    gemm(h16, qw16 + (size_t)l * HH, nullptr, nullptr, q16,  nullptr, TOK, cH, cH, 0);
    gemm(h16, kw16 + (size_t)l * HH, nullptr, nullptr, k16,  nullptr, TOK, cH, cH, 0);
    gemm(h16, vw16 + (size_t)l * HH, nullptr, nullptr, v16b, nullptr, TOK, cH, cH, 0);

    attn_flash_wmma<<<cB * cNH * (cS / 128), dim3(256), 0, stream>>>(q16, k16, v16b, a16);

    gemm(a16, ow16 + (size_t)l * HH, ob + (size_t)l * cH, xa, nullptr, xb, TOK, cH, cH, 0);
    { float* t = xa; xa = xb; xb = t; }

    ln_to_bf16<<<TOK, 256, 0, stream>>>(xa, ln2_w + (size_t)l * cH, ln2_b + (size_t)l * cH, h16);
    gemm(h16, w116 + (size_t)l * IH, b1 + (size_t)l * cI, nullptr, ff16, nullptr, TOK, cI, cH, 1);
    gemm(ff16, w216 + (size_t)l * IH, b2 + (size_t)l * cH, xa, nullptr, xb, TOK, cH, cI, 0);
    { float* t = xa; xa = xb; xb = t; }
  }

  ln_to_bf16<<<TOK, 256, 0, stream>>>(xa, lnf_w, lnf_b, h16);
  gemm(h16, lm16, nullptr, nullptr, nullptr, (float*)d_out, TOK, cV, cH, 0);
}
